// MultiHeadAttention_32684701122974
// MI455X (gfx1250) — matrix operand order repaired and checked by reading, not compiled, not run on hardware
//
#include <hip/hip_runtime.h>
#include <hip/hip_bf16.h>

typedef __bf16 bf16_t;
typedef __attribute__((ext_vector_type(16))) __bf16 v16bf;
typedef __attribute__((ext_vector_type(8)))  __bf16 v8bf;
typedef __attribute__((ext_vector_type(4)))  __bf16 v4bf;
typedef __attribute__((ext_vector_type(8)))  float  v8f;
typedef __attribute__((ext_vector_type(4)))  int    v4i;

#define MODEL_DIM 1024
#define N_HEADS   16
#define HEAD_DIM  64
#define BATCH     2
#define SEQ       2048
#define MTOT      (BATCH*SEQ)      // 4096 activation rows

// ---------------------------------------------------------------------------
// CDNA5 async global->LDS copy (ASYNCcnt path), with safe fallback
// ---------------------------------------------------------------------------
#if __has_builtin(__builtin_amdgcn_global_load_async_to_lds_b128)
#define HAS_ASYNC_LDS 1
#else
#define HAS_ASYNC_LDS 0
#endif

typedef __attribute__((address_space(1))) v4i* gptr_v4i;
typedef __attribute__((address_space(3))) v4i* lptr_v4i;

static __device__ __forceinline__ void async_copy16(const bf16_t* gsrc, bf16_t* lds_dst) {
#if HAS_ASYNC_LDS
  v4i* g = (v4i*)(void*)const_cast<bf16_t*>(gsrc);
  v4i* l = (v4i*)(void*)lds_dst;
  __builtin_amdgcn_global_load_async_to_lds_b128(
      (gptr_v4i)g, (lptr_v4i)l, /*offset=*/0, /*cpol=*/0);
#else
  *(v8bf*)lds_dst = *(const v8bf*)gsrc;
#endif
}

template <int N>
static __device__ __forceinline__ void wait_async() {
#if HAS_ASYNC_LDS
#if __has_builtin(__builtin_amdgcn_s_wait_asynccnt)
  __builtin_amdgcn_s_wait_asynccnt((short)N);
#else
  asm volatile("s_wait_asynccnt %0" :: "i"(N) : "memory");
#endif
#endif
}

// ---------------------------------------------------------------------------
// WMMA helpers (CDNA5 wave32, 16x16x32 bf16 -> f32)
// ---------------------------------------------------------------------------
static __device__ __forceinline__ v8f wmma_bf16(v16bf a, v16bf b, v8f c) {
  return __builtin_amdgcn_wmma_f32_16x16x32_bf16(
      /*neg_a=*/false, a, /*neg_b=*/false, b,
      /*c_mod=*/(short)0, c, /*reuse_a=*/false, /*reuse_b=*/false);
}

// A-matrix fragment: 16x32 bf16 tile, row-major source with leading dim lda.
// Lane L<16 -> row L, elements K in {0..7} u {16..23}; L>=16 -> +8 K offset.
static __device__ __forceinline__ v16bf load_a_frag(const bf16_t* base, int lda, int lane) {
  const int half = (lane >> 4) & 1;
  const int r    = lane & 15;
  const bf16_t* p = base + r * lda + 8 * half;
  v8bf c0 = *(const v8bf*)(p);
  v8bf c1 = *(const v8bf*)(p + 16);
  v16bf a;
#pragma unroll
  for (int e = 0; e < 8; ++e) { a[e] = c0[e]; a[e + 8] = c1[e]; }
  return a;
}

// B-matrix fragment: 32x16 tile given as Bt[n][k] (each output column n holds
// its K-run contiguously, leading dim ldb). Lane n=lane&15, K in {0..7} u {16..23}, +8 for lane>=16.
static __device__ __forceinline__ v16bf load_b_frag(const bf16_t* bt, int ldb, int lane) {
  const int khalf = (lane >> 4) & 1;
  const int n     = lane & 15;
  const bf16_t* p = bt + n * ldb + khalf * 8;
  v8bf c0 = *(const v8bf*)(p);
  v8bf c1 = *(const v8bf*)(p + 16);
  v16bf b;
#pragma unroll
  for (int e = 0; e < 8; ++e) { b[e] = c0[e]; b[e + 8] = c1[e]; }
  return b;
}

// ---------------------------------------------------------------------------
// f32 -> bf16 conversion (n must be a multiple of 4)
// ---------------------------------------------------------------------------
__global__ __launch_bounds__(256) void cvt_f32_bf16(const float* __restrict__ src,
                                                    bf16_t* __restrict__ dst, int n4) {
  int i = blockIdx.x * 256 + threadIdx.x;
  if (i < n4) {
    float4 v = ((const float4*)src)[i];
    v4bf o;
    o[0] = (bf16_t)v.x; o[1] = (bf16_t)v.y; o[2] = (bf16_t)v.z; o[3] = (bf16_t)v.w;
    ((v4bf*)dst)[i] = o;
  }
}

// ---------------------------------------------------------------------------
// GEMM: Y = X @ W^T + bias.  X:[MTOT x 1024] bf16 row-major, W:[1024 x 1024]
// bf16 row-major (row n = weights of output column n == Bt layout), bias f32.
// MODE 0: write bf16 into [B, H, S, 64] split-head layout.
// MODE 1: write f32 into [MTOT, 1024].
// Block: 256 threads (8 waves). C tile 128x128; wave grid 4x2; each wave owns
// 2 M-tiles x 4 N-tiles (8 WMMA accumulators).  Weight tile (128 N x 32 K) is
// double-buffered in LDS via async global->LDS copies (ASYNCcnt pipeline).
// ---------------------------------------------------------------------------
#define BSLD 40   // padded LDS leading dim for B tile (80B rows, 16B aligned)

template <int MODE>
__global__ __launch_bounds__(256) void gemm_bias_kernel(const bf16_t* __restrict__ X,
                                                        const bf16_t* __restrict__ W,
                                                        const float* __restrict__ bias,
                                                        void* __restrict__ out) {
  __shared__ bf16_t Bs[2][128][BSLD];

  const int lane = threadIdx.x & 31;
  const int wave = threadIdx.x >> 5;
  const int wrow = wave >> 1;              // 0..3  (M direction)
  const int wcol = wave & 1;               // 0..1  (N direction)
  const int block_n = blockIdx.y * 128;
  const int m0  = blockIdx.x * 128 + wrow * 32;
  const int wn0 = wcol * 64;               // wave's N offset inside the tile

  // stage a 128x32 slice of W (Bt layout) into LDS buffer `buf`
  auto issueB = [&](int k0, int buf) {
#pragma unroll
    for (int j = 0; j < 2; ++j) {
      const int idx = threadIdx.x * 2 + j;   // 0..511 chunks of 8 bf16
      const int row = idx >> 2;              // 0..127 (N)
      const int col = (idx & 3) * 8;         // 0..24  (K)
      async_copy16(W + (size_t)(block_n + row) * MODEL_DIM + k0 + col,
                   &Bs[buf][row][col]);
    }
  };

  v8f acc[2][4] = {};
  issueB(0, 0);

  for (int kk = 0; kk < MODEL_DIM / 32; ++kk) {
    const int k0 = kk * 32;
    if (kk + 1 < MODEL_DIM / 32) {
      issueB(k0 + 32, (kk + 1) & 1);
      wait_async<2>();                      // retire stage kk, keep kk+1 in flight
    } else {
      wait_async<0>();
    }
    __syncthreads();                        // publish stage kk to all waves

    v16bf a0 = load_a_frag(X + (size_t)m0 * MODEL_DIM + k0, MODEL_DIM, lane);
    v16bf a1 = load_a_frag(X + (size_t)(m0 + 16) * MODEL_DIM + k0, MODEL_DIM, lane);
    __builtin_prefetch(X + (size_t)m0 * MODEL_DIM + k0 + 32, 0, 0);
    __builtin_prefetch(X + (size_t)(m0 + 16) * MODEL_DIM + k0 + 32, 0, 0);

    const bf16_t* bs = &Bs[kk & 1][0][0];
#pragma unroll
    for (int t = 0; t < 4; ++t) {
      v16bf b = load_b_frag(bs + (size_t)(wn0 + t * 16) * BSLD, BSLD, lane);
      acc[0][t] = wmma_bf16(a0, b, acc[0][t]);
      acc[1][t] = wmma_bf16(a1, b, acc[1][t]);
    }
    __syncthreads();                        // readers done before buffer reuse
  }

  const int half = (lane >> 4) & 1;
  const int nl   = lane & 15;
#pragma unroll
  for (int mt = 0; mt < 2; ++mt) {
#pragma unroll
    for (int t = 0; t < 4; ++t) {
      const int ncol = block_n + wn0 + t * 16 + nl;
      const float bv = bias[ncol];
#pragma unroll
      for (int i = 0; i < 8; ++i) {
        const int row = m0 + mt * 16 + i + 8 * half;
        const float v = acc[mt][t][i] + bv;
        if (MODE == 0) {
          const int b  = row >> 11;        // / SEQ
          const int s  = row & (SEQ - 1);
          const int h  = ncol >> 6;        // / HEAD_DIM
          const int dh = ncol & (HEAD_DIM - 1);
          ((bf16_t*)out)[((((size_t)b * N_HEADS + h) * SEQ) + s) * HEAD_DIM + dh] = (bf16_t)v;
        } else {
          ((float*)out)[(size_t)row * MODEL_DIM + ncol] = v;
        }
      }
    }
  }
}

// ---------------------------------------------------------------------------
// Flash-style attention.  Qp/Kp/Vp: [B, H, S, 64] bf16.  Output written as
// bf16 [B, S, H*64] (heads re-concatenated) ready for the output projection.
// Block: 128 threads (4 waves). Each block: 64 query rows of one (b,h);
// each wave owns one 16-row M-tile end-to-end (online softmax in registers).
// K tile staged via async global->LDS; V staged transposed through VGPRs.
// ---------------------------------------------------------------------------
#define LDP 72   // padded LDS leading dim (multiple of 8 -> 16B aligned rows)

__global__ __launch_bounds__(128) void attention_kernel(const bf16_t* __restrict__ Qp,
                                                        const bf16_t* __restrict__ Kp,
                                                        const bf16_t* __restrict__ Vp,
                                                        const float* __restrict__ mask,
                                                        bf16_t* __restrict__ attn_out) {
  __shared__ bf16_t Kt[64][LDP];      // [key][dh]
  __shared__ bf16_t Vt[64][LDP];      // [dh][key]  (transposed V tile)
  __shared__ bf16_t Pt[4][16][LDP];   // per-wave P tile [q][key]

  const int lane = threadIdx.x & 31;
  const int wave = threadIdx.x >> 5;  // 0..3
  const int bh = blockIdx.y;
  const int b  = bh >> 4;             // / N_HEADS
  const int h  = bh & (N_HEADS - 1);

  const bf16_t* Qb = Qp + (size_t)bh * SEQ * HEAD_DIM;
  const bf16_t* Kb = Kp + (size_t)bh * SEQ * HEAD_DIM;
  const bf16_t* Vb = Vp + (size_t)bh * SEQ * HEAD_DIM;

  const int q0 = blockIdx.x * 64 + wave * 16;

  // Q fragments for this wave's 16 query rows (K dim = 64 -> 2 k-steps)
  v16bf qa[2];
  qa[0] = load_a_frag(Qb + (size_t)q0 * HEAD_DIM + 0, HEAD_DIM, lane);
  qa[1] = load_a_frag(Qb + (size_t)q0 * HEAD_DIM + 32, HEAD_DIM, lane);

  v8f o[4] = {};                      // output accumulators over dh (4 N-tiles)
  float m_run[8], l_run[8];
#pragma unroll
  for (int i = 0; i < 8; ++i) { m_run[i] = -3.0e38f; l_run[i] = 0.0f; }

  const int half = (lane >> 4) & 1;
  const int nl   = lane & 15;
  const float scale = 0.125f;         // 1/sqrt(64)

  for (int kt = 0; kt < SEQ; kt += 64) {
    __syncthreads();  // previous-iteration LDS consumers done before overwrite
    // 64x64 tile = 512 chunks of 8 bf16; 128 thr -> 4 passes.
    // K: async copy straight into LDS.  V: through VGPRs with transpose.
#pragma unroll
    for (int p = 0; p < 4; ++p) {
      const int idx = p * 128 + threadIdx.x;   // 0..511
      const int row = idx >> 3;                // 0..63 (key within tile)
      const int col = (idx & 7) * 8;           // 0..56 (dh)
      async_copy16(Kb + (size_t)(kt + row) * HEAD_DIM + col, &Kt[row][col]);
      v8bf dv = *(const v8bf*)(Vb + (size_t)(kt + row) * HEAD_DIM + col);
#pragma unroll
      for (int e = 0; e < 8; ++e) Vt[col + e][row] = dv[e];
    }
    wait_async<0>();
    __syncthreads();

    // scores: 16 q x 64 k  (4 N-tiles over keys)
    v8f s[4] = {};
#pragma unroll
    for (int ks = 0; ks < 2; ++ks) {
#pragma unroll
      for (int t = 0; t < 4; ++t) {
        v16bf bk = load_b_frag(&Kt[t * 16][0] + ks * 32, LDP, lane);
        s[t] = wmma_bf16(qa[ks], bk, s[t]);
      }
    }

    // scale + mask + online softmax
    float pv[4][8];
    float tmax[8];
#pragma unroll
    for (int i = 0; i < 8; ++i) tmax[i] = -3.0e38f;
#pragma unroll
    for (int t = 0; t < 4; ++t) {
      const int key = kt + t * 16 + nl;
      const float mb = (1.0f - mask[(size_t)b * SEQ + key]) * -1.0e30f;
#pragma unroll
      for (int i = 0; i < 8; ++i) {
        const float v = s[t][i] * scale + mb;
        pv[t][i] = v;
        tmax[i] = fmaxf(tmax[i], v);
      }
    }
    // row max across the 16 lanes of each half-group (wave32 shuffles)
#pragma unroll
    for (int off = 1; off < 16; off <<= 1)
#pragma unroll
      for (int i = 0; i < 8; ++i)
        tmax[i] = fmaxf(tmax[i], __shfl_xor(tmax[i], off, 32));

    float corr[8];
#pragma unroll
    for (int i = 0; i < 8; ++i) {
      const float nm = fmaxf(m_run[i], tmax[i]);
      corr[i] = __expf(m_run[i] - nm);
      m_run[i] = nm;
    }
    float rsum[8];
#pragma unroll
    for (int i = 0; i < 8; ++i) rsum[i] = 0.0f;
#pragma unroll
    for (int t = 0; t < 4; ++t)
#pragma unroll
      for (int i = 0; i < 8; ++i) {
        const float e = __expf(pv[t][i] - m_run[i]);
        pv[t][i] = e;
        rsum[i] += e;
      }
#pragma unroll
    for (int off = 1; off < 16; off <<= 1)
#pragma unroll
      for (int i = 0; i < 8; ++i)
        rsum[i] += __shfl_xor(rsum[i], off, 32);
#pragma unroll
    for (int i = 0; i < 8; ++i) l_run[i] = l_run[i] * corr[i] + rsum[i];
#pragma unroll
    for (int t = 0; t < 4; ++t)
#pragma unroll
      for (int i = 0; i < 8; ++i) o[t][i] *= corr[i];

    // write P (C-layout) to wave-private LDS as bf16, then re-read as A-frags
#pragma unroll
    for (int t = 0; t < 4; ++t)
#pragma unroll
      for (int i = 0; i < 8; ++i)
        Pt[wave][i + 8 * half][t * 16 + nl] = (bf16_t)pv[t][i];

    // O += P @ V   (K dim = 64 keys -> 2 k-steps)
#pragma unroll
    for (int ks = 0; ks < 2; ++ks) {
      v16bf pa = load_a_frag(&Pt[wave][0][0] + ks * 32, LDP, lane);
#pragma unroll
      for (int t = 0; t < 4; ++t) {
        v16bf bv = load_b_frag(&Vt[t * 16][0] + ks * 32, LDP, lane);
        o[t] = wmma_bf16(pa, bv, o[t]);
      }
    }
  }

  // epilogue: O / l, store bf16 into [B, S, H*64]
#pragma unroll
  for (int t = 0; t < 4; ++t) {
#pragma unroll
    for (int i = 0; i < 8; ++i) {
      const int s_idx = q0 + i + 8 * half;
      const int dh = t * 16 + nl;
      const float v = o[t][i] / l_run[i];
      attn_out[((size_t)b * SEQ + s_idx) * MODEL_DIM + h * HEAD_DIM + dh] = (bf16_t)v;
    }
  }
}

// ---------------------------------------------------------------------------
// Host-side launch.  Input dict order: k,q,v,attn_mask,Wq,bq,Wk,bk,Wv,bv,Wo,bo
// ---------------------------------------------------------------------------
extern "C" void kernel_launch(void* const* d_in, const int* in_sizes, int n_in,
                              void* d_out, int out_size, void* d_ws, size_t ws_size,
                              hipStream_t stream) {
  const float* k_f  = (const float*)d_in[0];
  const float* q_f  = (const float*)d_in[1];
  const float* v_f  = (const float*)d_in[2];
  const float* mask = (const float*)d_in[3];
  const float* Wq_f = (const float*)d_in[4];
  const float* bq   = (const float*)d_in[5];
  const float* Wk_f = (const float*)d_in[6];
  const float* bk   = (const float*)d_in[7];
  const float* Wv_f = (const float*)d_in[8];
  const float* bv   = (const float*)d_in[9];
  const float* Wo_f = (const float*)d_in[10];
  const float* bo   = (const float*)d_in[11];

  const size_t ACT = (size_t)MTOT * MODEL_DIM;      // 4,194,304 elements
  const size_t WEL = (size_t)MODEL_DIM * MODEL_DIM; // 1,048,576 elements

  bf16_t* qb    = (bf16_t*)d_ws;    // activations, bf16
  bf16_t* kb    = qb + ACT;
  bf16_t* vb    = kb + ACT;
  bf16_t* Wqb   = vb + ACT;         // weights, bf16
  bf16_t* Wkb   = Wqb + WEL;
  bf16_t* Wvb   = Wkb + WEL;
  bf16_t* Wob   = Wvb + WEL;
  bf16_t* Qp    = Wob + WEL;        // projected, [B,H,S,64] bf16
  bf16_t* Kp    = Qp + ACT;
  bf16_t* Vp    = Kp + ACT;
  bf16_t* attnb = Vp + ACT;         // attention output, [B,S,1024] bf16
  // total: 33,554,432 bf16 elements = 64 MB of workspace

  // 1) f32 -> bf16 conversions
  {
    const int n4a = (int)(ACT / 4), n4w = (int)(WEL / 4);
    dim3 blk(256);
    cvt_f32_bf16<<<dim3((n4a + 255) / 256), blk, 0, stream>>>(q_f,  qb,  n4a);
    cvt_f32_bf16<<<dim3((n4a + 255) / 256), blk, 0, stream>>>(k_f,  kb,  n4a);
    cvt_f32_bf16<<<dim3((n4a + 255) / 256), blk, 0, stream>>>(v_f,  vb,  n4a);
    cvt_f32_bf16<<<dim3((n4w + 255) / 256), blk, 0, stream>>>(Wq_f, Wqb, n4w);
    cvt_f32_bf16<<<dim3((n4w + 255) / 256), blk, 0, stream>>>(Wk_f, Wkb, n4w);
    cvt_f32_bf16<<<dim3((n4w + 255) / 256), blk, 0, stream>>>(Wv_f, Wvb, n4w);
    cvt_f32_bf16<<<dim3((n4w + 255) / 256), blk, 0, stream>>>(Wo_f, Wob, n4w);
  }

  // 2) Q/K/V projections (bf16 WMMA, split-head bf16 output)
  {
    dim3 grid(MTOT / 128, MODEL_DIM / 128), blk(256);
    gemm_bias_kernel<0><<<grid, blk, 0, stream>>>(qb, Wqb, bq, (void*)Qp);
    gemm_bias_kernel<0><<<grid, blk, 0, stream>>>(kb, Wkb, bk, (void*)Kp);
    gemm_bias_kernel<0><<<grid, blk, 0, stream>>>(vb, Wvb, bv, (void*)Vp);
  }

  // 3) flash attention
  {
    dim3 grid(SEQ / 64, BATCH * N_HEADS), blk(128);
    attention_kernel<<<grid, blk, 0, stream>>>(Qp, Kp, Vp, mask, attnb);
  }

  // 4) output projection (f32 output + bias)
  {
    dim3 grid(MTOT / 128, MODEL_DIM / 128), blk(256);
    gemm_bias_kernel<1><<<grid, blk, 0, stream>>>(attnb, Wob, bo, d_out);
  }
}
